// CrossAttn_79362405695994
// MI455X (gfx1250) — matrix operand order repaired and checked by reading, not compiled, not run on hardware
//
#include <hip/hip_runtime.h>
#include <hip/hip_bf16.h>

// ---------------------------------------------------------------------------
// CDNA5 (gfx1250) cross-attention pipeline, all GEMMs via v_wmma_f32_16x16x32_bf16
// ---------------------------------------------------------------------------

typedef __attribute__((ext_vector_type(16))) __bf16         v16bf;
typedef __attribute__((ext_vector_type(8)))  float          v8f;
typedef __attribute__((ext_vector_type(8)))  unsigned short v8us;
typedef int v4i __attribute__((vector_size(16)));           // int4 for async builtin

union FragBf {            // 16 bf16 values = 8 VGPRs; filled by two 16B LDS reads
    v16bf v;
    v8us  h[2];
};

// float -> bf16 bits, round-to-nearest-even
__device__ __forceinline__ unsigned short f2bf(float f) {
    unsigned u = __builtin_bit_cast(unsigned, f);
    u += 0x7fffu + ((u >> 16) & 1u);
    return (unsigned short)(u >> 16);
}

// ---- gfx1250 async global->LDS staging (ASYNCcnt), with safe fallback ------
#if defined(__gfx1250__) && __has_builtin(__builtin_amdgcn_global_load_async_to_lds_b128)
#define HAVE_ASYNC_LDS 1
#else
#define HAVE_ASYNC_LDS 0
#endif

__device__ __forceinline__ void asyncCopy16B(const unsigned short* g, unsigned short* l) {
#if HAVE_ASYNC_LDS
    __builtin_amdgcn_global_load_async_to_lds_b128(
        (__attribute__((address_space(1))) v4i*)g,
        (__attribute__((address_space(3))) v4i*)l,
        0, 0);
#else
    *(v8us*)l = *(const v8us*)g;
#endif
}
__device__ __forceinline__ void waitAsync() {
#if HAVE_ASYNC_LDS
#if __has_builtin(__builtin_amdgcn_s_wait_asynccnt)
    __builtin_amdgcn_s_wait_asynccnt(0);
#else
    asm volatile("s_wait_asynccnt 0" ::: "memory");
#endif
#endif
}

// ---------------------------------------------------------------------------
// Generic batched strided GEMM:  C[b1,b2,m,n] = sum_k A[k,m] * B[k,n]
// A,B are bf16 bits (ushort), C is f32 or bf16 bits.
// Tile: 128x64 per block (256 threads = 8 waves, each 32x32), K step 32.
// PATH 0: A and B k-contiguous: async b128 global->LDS (or b128+b128 fallback).
// PATH 1: A m-contiguous, B n-contiguous: b128 global + b16 LDS scatter
//         (transpose into canonical [free][k] tiles).
// ---------------------------------------------------------------------------
#define TM 128
#define TN 64
#define TK 32
#define LDP (TK + 8)      // LDS row pitch in elements (80 bytes, keeps 16B align)

template <int PATH, bool OUTBF>
__global__ __launch_bounds__(256) void gemm_bf16_wmma(
    const unsigned short* __restrict__ Ag,
    const unsigned short* __restrict__ Bg,
    void* __restrict__ Cg,
    int K,
    long long sAm, long long sAk,
    long long sBn, long long sBk,
    long long sCm, long long sCn,
    int B2,
    long long sA1, long long sA2,
    long long sB1, long long sB2,
    long long sC1, long long sC2)
{
    __shared__ unsigned short As[TM][LDP];   // (m, k)
    __shared__ unsigned short Bs[TN][LDP];   // transposed (n, k)

    const int tid  = threadIdx.x;
    const int wave = tid >> 5;
    const int lane = tid & 31;
    const int half = lane >> 4;     // 0: lanes 0-15, 1: lanes 16-31
    const int l16  = lane & 15;

    const int n0 = blockIdx.x * TN;
    const int m0 = blockIdx.y * TM;
    const int z  = blockIdx.z;
    const int b1 = z / B2;
    const int b2 = z - b1 * B2;

    const unsigned short* __restrict__ Ab = Ag + b1 * sA1 + b2 * sA2;
    const unsigned short* __restrict__ Bb = Bg + b1 * sB1 + b2 * sB2;
    const long long cOff = b1 * sC1 + b2 * sC2;

    const int wm = (wave >> 1) * 32;   // 8 waves: 4 (rows) x 2 (cols) of 32x32
    const int wn = (wave & 1) * 32;

    // ---- per-thread loop-invariant staging pointers (bumped per k-step) ----
    const unsigned short* __restrict__ aPtr0;  // A: 512 vectors of 8; 2 per thread
    const unsigned short* __restrict__ aPtr1;
    unsigned short* aLds0;
    unsigned short* aLds1;
    const unsigned short* __restrict__ bPtr;   // B: 256 vectors; 1 per thread
    unsigned short* bLds;
    long long aStep, bStep;

    if constexpr (PATH == 0) {
        {
            const int m = tid >> 2, kv = tid & 3;          // 4 vectors per row
            aPtr0 = Ab + (long long)(m0 + m) * sAm + kv * 8;
            aLds0 = &As[m][kv * 8];
        }
        {
            const int v = tid + 256;
            const int m = v >> 2, kv = v & 3;
            aPtr1 = Ab + (long long)(m0 + m) * sAm + kv * 8;
            aLds1 = &As[m][kv * 8];
        }
        const int n = tid >> 2, kv = tid & 3;
        bPtr = Bb + (long long)(n0 + n) * sBn + kv * 8;
        bLds = &Bs[n][kv * 8];
        aStep = TK;
        bStep = TK;
    } else {
        {
            const int k = tid & 31, mg = tid >> 5;         // 16 m-groups of 8
            aPtr0 = Ab + (m0 + mg * 8) + (long long)k * sAk;
            aLds0 = &As[mg * 8][k];
        }
        {
            const int v = tid + 256;
            const int k = v & 31, mg = v >> 5;
            aPtr1 = Ab + (m0 + mg * 8) + (long long)k * sAk;
            aLds1 = &As[mg * 8][k];
        }
        const int k = tid & 31, ng = tid >> 5;
        bPtr = Bb + (n0 + ng * 8) + (long long)k * sBk;
        bLds = &Bs[ng * 8][k];
        aStep = (long long)TK * sAk;
        bStep = (long long)TK * sBk;
    }

    v8f acc[2][2];
#pragma unroll
    for (int t = 0; t < 2; ++t)
#pragma unroll
        for (int u = 0; u < 2; ++u)
#pragma unroll
            for (int r = 0; r < 8; ++r) acc[t][u][r] = 0.0f;

    for (int k0 = 0; k0 < K; k0 += TK) {
        // ---------------- stage A (128x32) and B (64x32, transposed) ----------
        if constexpr (PATH == 0) {
            asyncCopy16B(aPtr0, aLds0);        // global_load_async_to_lds_b128
            asyncCopy16B(aPtr1, aLds1);        // (or b128 load+store fallback)
            asyncCopy16B(bPtr,  bLds);
        } else {
            const v8us d0 = *(const v8us*)aPtr0;
            const v8us d1 = *(const v8us*)aPtr1;
            const v8us db = *(const v8us*)bPtr;
#pragma unroll
            for (int e = 0; e < 8; ++e) aLds0[e * LDP] = d0[e];
#pragma unroll
            for (int e = 0; e < 8; ++e) aLds1[e * LDP] = d1[e];
#pragma unroll
            for (int e = 0; e < 8; ++e) bLds[e * LDP] = db[e];
        }
        aPtr0 += aStep; aPtr1 += aStep; bPtr += bStep;

        // prefetch next K tile (global_prefetch_b8)
        if (k0 + TK < K) {
            __builtin_prefetch(aPtr0, 0, 1);
            __builtin_prefetch(bPtr, 0, 1);
        }
        if constexpr (PATH == 0) waitAsync();
        __syncthreads();

        // ---- load fragments per ISA 16-bit A(16x32) / B(32x16) layouts ----
        FragBf a[2], b[2];
#pragma unroll
        for (int t = 0; t < 2; ++t) {
            const int row = wm + 16 * t + l16;           // A: lane holds row M=l16
            a[t].h[0] = *(const v8us*)&As[row][8 * half];        // K = 8*half + 0..7
            a[t].h[1] = *(const v8us*)&As[row][16 + 8 * half];   // K = 16+8*half + 0..7
        }
#pragma unroll
        for (int u = 0; u < 2; ++u) {
            const int col = wn + 16 * u + l16;           // B: lane holds col N=l16
            b[u].h[0] = *(const v8us*)&Bs[col][8 * half];        // K = 8*half + 0..7
            b[u].h[1] = *(const v8us*)&Bs[col][16 + 8 * half];   // K = 16+8*half + 0..7
        }
#pragma unroll
        for (int t = 0; t < 2; ++t)
#pragma unroll
            for (int u = 0; u < 2; ++u)
                acc[t][u] = __builtin_amdgcn_wmma_f32_16x16x32_bf16(
                    false, a[t].v, false, b[u].v, (short)0, acc[t][u], false, false);
        __syncthreads();
    }

    // ---- store D: element r of lane -> (m = r + 8*half, n = l16) ----
#pragma unroll
    for (int t = 0; t < 2; ++t)
#pragma unroll
        for (int u = 0; u < 2; ++u) {
            const int nn = n0 + wn + 16 * u + l16;
#pragma unroll
            for (int r = 0; r < 8; ++r) {
                const int mm = m0 + wm + 16 * t + r + 8 * half;
                const long long idx = cOff + (long long)mm * sCm + (long long)nn * sCn;
                const float v = acc[t][u][r];
                if constexpr (OUTBF) ((unsigned short*)Cg)[idx] = f2bf(v);
                else                 ((float*)Cg)[idx] = v;
            }
        }
}

// ---------------------------------------------------------------------------
// fp32 -> bf16-bits conversion (grid-stride)
// ---------------------------------------------------------------------------
__global__ __launch_bounds__(256) void cvt_f32_bf16(
    const float* __restrict__ x, unsigned short* __restrict__ y, long long n)
{
    long long i = (long long)blockIdx.x * blockDim.x + threadIdx.x;
    const long long step = (long long)gridDim.x * blockDim.x;
    for (; i < n; i += step) y[i] = f2bf(x[i]);
}

// ---------------------------------------------------------------------------
// Instance-norm stats: one block per batch item; stats[2i]=mu, stats[2i+1]=rstd
// ---------------------------------------------------------------------------
__global__ __launch_bounds__(256) void inorm_stats(
    const float* __restrict__ x, float* __restrict__ stats, long long per)
{
    __shared__ float shs[256];
    __shared__ float shq[256];
    const long long base = (long long)blockIdx.x * per;
    float s = 0.0f, q = 0.0f;
    for (long long i = threadIdx.x; i < per; i += 256) {
        const float v = x[base + i];
        s += v;
        q += v * v;
    }
    const int tid = threadIdx.x;
    shs[tid] = s; shq[tid] = q;
    __syncthreads();
    for (int off = 128; off > 0; off >>= 1) {
        if (tid < off) { shs[tid] += shs[tid + off]; shq[tid] += shq[tid + off]; }
        __syncthreads();
    }
    if (tid == 0) {
        const float inv = 1.0f / (float)per;
        const float mu  = shs[0] * inv;
        const float var = shq[0] * inv - mu * mu;
        stats[2 * blockIdx.x]     = mu;
        stats[2 * blockIdx.x + 1] = rsqrtf(var + 1e-5f);
    }
}

// ---------------------------------------------------------------------------
// Fused instance-norm + row softmax, bf16 output. One block per row.
// ---------------------------------------------------------------------------
__global__ __launch_bounds__(256) void inorm_softmax(
    const float* __restrict__ x, unsigned short* __restrict__ y,
    const float* __restrict__ stats, int cols, int rowsPerBatch)
{
    __shared__ float sh[256];
    const int row   = blockIdx.x;
    const int batch = row / rowsPerBatch;
    const float mu   = stats[2 * batch];
    const float rstd = stats[2 * batch + 1];
    const long long base = (long long)row * cols;
    const int tid = threadIdx.x;

    float mx = -3.4e38f;
    for (int i = tid; i < cols; i += 256)
        mx = fmaxf(mx, (x[base + i] - mu) * rstd);
    sh[tid] = mx; __syncthreads();
    for (int off = 128; off > 0; off >>= 1) {
        if (tid < off) sh[tid] = fmaxf(sh[tid], sh[tid + off]);
        __syncthreads();
    }
    mx = sh[0]; __syncthreads();

    float s = 0.0f;
    for (int i = tid; i < cols; i += 256)
        s += __expf((x[base + i] - mu) * rstd - mx);
    sh[tid] = s; __syncthreads();
    for (int off = 128; off > 0; off >>= 1) {
        if (tid < off) sh[tid] += sh[tid + off];
        __syncthreads();
    }
    const float inv = 1.0f / sh[0];
    for (int i = tid; i < cols; i += 256)
        y[base + i] = f2bf(__expf((x[base + i] - mu) * rstd - mx) * inv);
}

// ---------------------------------------------------------------------------
// Host orchestration
// ---------------------------------------------------------------------------
extern "C" void kernel_launch(void* const* d_in, const int* in_sizes, int n_in,
                              void* d_out, int out_size, void* d_ws, size_t ws_size,
                              hipStream_t stream)
{
    (void)in_sizes; (void)n_in; (void)out_size; (void)ws_size;

    // Problem constants
    constexpr long long NN = 4096, CC = 128, HH = 8, HC = 1024, BB = 8, bL = 4;
    constexpr long long EMB_ELEMS = BB * NN * CC;        // 4,194,304
    constexpr long long EMB_U_OFF = bL * NN * CC;        // 2,097,152 (upper half)
    constexpr long long W_ELEMS   = HC * CC;             // 131,072

    // Workspace layout (byte offsets, 256-aligned)
    char* ws = (char*)d_ws;
    unsigned short* embBf = (unsigned short*)(ws + 0);                    //  8.39 MB
    unsigned short* wBf   = (unsigned short*)(ws + 8388608LL);            //  2.10 MB (8 weights)
    unsigned short* P0    = (unsigned short*)(ws + 10485760LL);           // 33.55 MB  Q  / Q2
    unsigned short* P1    = (unsigned short*)(ws + 44040192LL);           // 33.55 MB  K  / K2(T)
    unsigned short* P2    = (unsigned short*)(ws + 77594624LL);           // 33.55 MB  V  / V2(T)
    float*          S     = (float*)         (ws + 111149056LL);          // 67.11 MB  attn / attn2
    unsigned short* SIM   = (unsigned short*)(ws + 178257920LL);          // 33.55 MB  sim / sim2
    unsigned short* CTX   = (unsigned short*)(ws + 211812352LL);          // 33.55 MB  ctx / ctx2
    float*          ST1   = (float*)         (ws + 245366784LL);          // 32 pairs
    float*          ST2   = ST1 + 64;                                     //  4 pairs

    const unsigned short* embU = embBf + EMB_U_OFF;
    const unsigned short* w_q_uu = wBf + 0 * W_ELEMS;
    const unsigned short* w_k_uu = wBf + 1 * W_ELEMS;
    const unsigned short* w_v_uu = wBf + 2 * W_ELEMS;
    const unsigned short* w_q_lu = wBf + 3 * W_ELEMS;
    const unsigned short* w_k_lu = wBf + 4 * W_ELEMS;
    const unsigned short* w_v_lu = wBf + 5 * W_ELEMS;
    const unsigned short* w_o_u  = wBf + 6 * W_ELEMS;
    const unsigned short* w_o_l  = wBf + 7 * W_ELEMS;

    float* outL = (float*)d_out;                       // O_l -> batches 0..3
    float* outU = (float*)d_out + bL * NN * CC;        // O_u -> batches 4..7

    // PATH 0: sAk==1 && sBk==1 (k-contiguous), PATH 1: sAm==1 && sBn==1.
    auto gemmKK = [&](const unsigned short* A, const unsigned short* B, void* C, int outBf,
                      int M, int N, int K,
                      long long sAm, long long sBn,
                      long long sCm, long long sCn,
                      int B1, int B2,
                      long long sA1, long long sA2, long long sB1, long long sB2,
                      long long sC1, long long sC2) {
        dim3 grid(N / TN, M / TM, B1 * B2);
        if (outBf)
            gemm_bf16_wmma<0, true><<<grid, 256, 0, stream>>>(A, B, C, K,
                sAm, 1, sBn, 1, sCm, sCn, B2, sA1, sA2, sB1, sB2, sC1, sC2);
        else
            gemm_bf16_wmma<0, false><<<grid, 256, 0, stream>>>(A, B, C, K,
                sAm, 1, sBn, 1, sCm, sCn, B2, sA1, sA2, sB1, sB2, sC1, sC2);
    };
    auto gemmMN = [&](const unsigned short* A, const unsigned short* B, void* C,
                      int M, int N, int K,
                      long long sAk, long long sBk,
                      long long sCm, long long sCn,
                      int B1, int B2,
                      long long sA1, long long sA2, long long sB1, long long sB2,
                      long long sC1, long long sC2) {
        dim3 grid(N / TN, M / TM, B1 * B2);
        gemm_bf16_wmma<1, false><<<grid, 256, 0, stream>>>(A, B, C, K,
            1, sAk, 1, sBk, sCm, sCn, B2, sA1, sA2, sB1, sB2, sC1, sC2);
    };

    // ---- 0) fp32 -> bf16 conversions ----
    cvt_f32_bf16<<<2048, 256, 0, stream>>>((const float*)d_in[0], embBf, EMB_ELEMS);
    for (int w = 0; w < 8; ++w)
        cvt_f32_bf16<<<128, 256, 0, stream>>>((const float*)d_in[1 + w],
                                              wBf + (long long)w * W_ELEMS, W_ELEMS);

    // =================== Branch 1 (upper, emb[4:]) ===================
    // Q/K/V projections: (16384 x 1024) = embU(16384x128) @ W(1024x128)^T, bf16 out
    gemmKK(embU, w_q_uu, P0, 1, 16384, 1024, 128, 128, 128, 1024, 1,
           1, 1, 0, 0, 0, 0, 0, 0);
    gemmKK(embU, w_k_uu, P1, 1, 16384, 1024, 128, 128, 128, 1024, 1,
           1, 1, 0, 0, 0, 0, 0, 0);
    gemmKK(embU, w_v_uu, P2, 1, 16384, 1024, 128, 128, 128, 1024, 1,
           1, 1, 0, 0, 0, 0, 0, 0);

    // attn[b,h,c,d] = sum_n Q[b,n,h*128+c] * K[b,n,h*128+d]   (f32 out)
    gemmMN(P0, P1, S, 128, 128, 4096,
           /*sAk*/ 1024, /*sBk*/ 1024, /*C*/ 128, 1,
           4, 8, /*A batch*/ NN * HC, 128, /*B batch*/ NN * HC, 128,
           /*C batch*/ HH * 128 * 128, 128 * 128);

    // instance-norm (over 128x128 per (b,h)) + softmax rows of 128
    inorm_stats<<<32, 256, 0, stream>>>(S, ST1, 128 * 128);
    inorm_softmax<<<32 * 128, 256, 0, stream>>>(S, SIM, ST1, 128, 128);

    // ctx[b,h,c,n] = sum_d sim[b,h,c,d] * V[b,n,h*128+d]; stored as ctxr[b,n,c*8+h] (bf16)
    gemmKK(SIM, P2, CTX, 1, 128, 4096, 128,
           /*sAm*/ 128, /*sBn*/ 1024, /*C*/ 8, 1024,
           4, 8, /*A batch*/ HH * 128 * 128, 128 * 128, /*B batch*/ NN * HC, 128,
           /*C batch*/ NN * HC, 1);

    // O_u = ctxr @ wo_u^T  -> d_out upper half (f32)
    gemmKK(CTX, w_o_u, outU, 0, 16384, 128, 1024, 1024, 1024, 128, 1,
           1, 1, 0, 0, 0, 0, 0, 0);

    // =================== Branch 2 (lower) ===================
    // Q2 = emb_l @ wq_lu^T -> P0 (b,n,hc) contiguous (bf16)
    gemmKK(embBf, w_q_lu, P0, 1, 16384, 1024, 128, 128, 128, 1024, 1,
           1, 1, 0, 0, 0, 0, 0, 0);
    // K2[n, b*1024+hc] = (emb_u @ wk_lu^T)[b,n,hc]  -> transposed store into P1
    gemmKK(embU, w_k_lu, P1, 1, 4096, 1024, 128, 128, 128,
           /*Cm(n)*/ 4096, /*Cn(hc)*/ 1,
           4, 1, /*A batch*/ NN * CC, 0, 0, 0, /*C batch*/ 1024, 0);
    // V2 likewise -> P2
    gemmKK(embU, w_v_lu, P2, 1, 4096, 1024, 128, 128, 128, 4096, 1,
           4, 1, NN * CC, 0, 0, 0, 1024, 0);

    // attn2[b,c,d] = sum_n Q2[b,n,c] * K2[n,d]   (f32, 4 x 1024 x 4096)
    gemmMN(P0, P1, S, 1024, 4096, 4096,
           /*sAk*/ 1024, /*sBk*/ 4096, /*C*/ 4096, 1,
           4, 1, /*A batch*/ NN * HC, 0, 0, 0, /*C batch*/ HC * NN, 0);

    // instance-norm (over 1024x4096 per b) + softmax rows of 4096
    inorm_stats<<<4, 256, 0, stream>>>(S, ST2, 1024LL * 4096LL);
    inorm_softmax<<<4 * 1024, 256, 0, stream>>>(S, SIM, ST2, 4096, 1024);

    // ctx2[b,n,c] = sum_d sim2[b,c,d] * V2[n,d]  -> CTX (b,n,c) bf16
    gemmKK(SIM, P2, CTX, 1, 1024, 4096, 4096,
           /*sAm*/ 4096, /*sBn*/ 4096, /*C*/ 1, 1024,
           4, 1, /*A batch*/ HC * NN, 0, 0, 0, /*C batch*/ NN * HC, 0);

    // O_l = ctx2 @ wo_l^T -> d_out lower half (f32)
    gemmKK(CTX, w_o_l, outL, 0, 16384, 128, 1024, 1024, 1024, 128, 1,
           1, 1, 0, 0, 0, 0, 0, 0);
}
